// WIKG_MIL_20160576488026
// MI455X (gfx1250) — compile-verified, matrix operand order checked by reading, not run on hardware
//
#include <hip/hip_runtime.h>

// ---------------- types for WMMA ----------------
typedef __attribute__((ext_vector_type(16))) __bf16 v16bf;
typedef __attribute__((ext_vector_type(8)))  float  v8f;

union FragU {
    uint4 u[2];
    v16bf v;
};

__device__ __forceinline__ unsigned short f2bf(float f) {
    unsigned int u = __builtin_bit_cast(unsigned int, f);
    unsigned int r = u + 0x7FFFu + ((u >> 16) & 1u);   // RNE
    return (unsigned short)(r >> 16);
}

// ---------------- elementwise fp32 -> bf16 ----------------
__global__ void cvt_f32_bf16(const float* __restrict__ in,
                             unsigned short* __restrict__ out, int n) {
    int i = blockIdx.x * blockDim.x + threadIdx.x;
    if (i < n) out[i] = f2bf(in[i]);
}

// ---------------- generic WMMA GEMM:  out = epi(A[M,K] @ B[N,K]^T + bias) ----
// flags: 1 = leaky-relu, 2 = accumulate into existing out
#define GF_LRELU 1
#define GF_ACC   2

__global__ __launch_bounds__(256)
void gemm_bf16_wmma(const unsigned short* __restrict__ A,
                    const unsigned short* __restrict__ B,
                    const float* __restrict__ bias,
                    float* __restrict__ out,
                    unsigned short* __restrict__ out_bf, float bf_scale,
                    int M, int N, int K, int flags) {
    __shared__ __align__(16) unsigned short As[64][32];
    __shared__ __align__(16) unsigned short Bs[64][32];

    const int t    = threadIdx.x;
    const int wave = t >> 5;
    const int lane = t & 31;
    const int wm   = wave >> 1;       // 0..3 : 16-row sub-tile
    const int wn   = wave & 1;        // 0..1 : 32-col sub-tile
    const int half = lane >> 4;
    const int r16  = lane & 15;

    const int m0 = blockIdx.y * 64;
    const int n0 = blockIdx.x * 64;

    const int lr = t >> 2;            // 0..63 cooperative-load row
    const int lc = (t & 3) * 8;       // 0,8,16,24

    v8f acc0 = {};
    v8f acc1 = {};

    for (int k0 = 0; k0 < K; k0 += 32) {
        *(uint4*)&As[lr][lc] = *(const uint4*)&A[(size_t)(m0 + lr) * K + k0 + lc];
        *(uint4*)&Bs[lr][lc] = *(const uint4*)&B[(size_t)(n0 + lr) * K + k0 + lc];
        if (k0 + 32 < K) {
            __builtin_prefetch(&A[(size_t)(m0 + lr) * K + k0 + 32 + lc], 0, 1);
            __builtin_prefetch(&B[(size_t)(n0 + lr) * K + k0 + 32 + lc], 0, 1);
        }
        __syncthreads();

        FragU a, b0, b1;
        const int ar = wm * 16 + r16;
        a.u[0]  = *(const uint4*)&As[ar][half * 8];
        a.u[1]  = *(const uint4*)&As[ar][16 + half * 8];
        const int br = wn * 32 + r16;
        b0.u[0] = *(const uint4*)&Bs[br][half * 8];
        b0.u[1] = *(const uint4*)&Bs[br][16 + half * 8];
        b1.u[0] = *(const uint4*)&Bs[br + 16][half * 8];
        b1.u[1] = *(const uint4*)&Bs[br + 16][16 + half * 8];

        acc0 = __builtin_amdgcn_wmma_f32_16x16x32_bf16(false, a.v, false, b0.v,
                                                       (short)0, acc0, false, false);
        acc1 = __builtin_amdgcn_wmma_f32_16x16x32_bf16(false, a.v, false, b1.v,
                                                       (short)0, acc1, false, false);
        __syncthreads();
    }

    const int col0 = n0 + wn * 32 + r16;
    const int col1 = col0 + 16;
#pragma unroll
    for (int v = 0; v < 8; ++v) {
        const int row = m0 + wm * 16 + v + 8 * half;
        float e0 = acc0[v] + (bias ? bias[col0] : 0.f);
        float e1 = acc1[v] + (bias ? bias[col1] : 0.f);
        if (flags & GF_LRELU) {
            e0 = e0 > 0.f ? e0 : 0.01f * e0;
            e1 = e1 > 0.f ? e1 : 0.01f * e1;
        }
        size_t i0 = (size_t)row * N + col0;
        size_t i1 = (size_t)row * N + col1;
        if (flags & GF_ACC) { e0 += out[i0]; e1 += out[i1]; }
        out[i0] = e0;
        out[i1] = e1;
        if (out_bf) {
            out_bf[i0] = f2bf(e0 * bf_scale);
            out_bf[i1] = f2bf(e1 * bf_scale);
        }
    }
}

// ---------------- column mean over N (two stage) ----------------
__global__ void colsum_partial(const float* __restrict__ h, float* __restrict__ part) {
    // 32 blocks, 256 threads; block b owns rows [b*256, b*256+256), thread owns 2 cols
    const int b = blockIdx.x, t = threadIdx.x;
    float s0 = 0.f, s1 = 0.f;
    for (int r = 0; r < 256; ++r) {
        const float* row = h + (size_t)(b * 256 + r) * 512;
        s0 += row[t];
        s1 += row[t + 256];
    }
    part[b * 512 + t]       = s0;
    part[b * 512 + t + 256] = s1;
}

__global__ void colmean_final(const float* __restrict__ part, float* __restrict__ mean) {
    const int d = threadIdx.x;   // 512 threads
    float s = 0.f;
    for (int b = 0; b < 32; ++b) s += part[b * 512 + d];
    mean[d] = s * (1.0f / 8192.0f);
}

__global__ void mix_mean_bf16(const float* __restrict__ h, const float* __restrict__ mean,
                              unsigned short* __restrict__ hbf, int n) {
    int i = blockIdx.x * blockDim.x + threadIdx.x;
    if (i < n) hbf[i] = f2bf((h[i] + mean[i & 511]) * 0.5f);
}

// ---------------- streaming attention + top-6 ----------------
__global__ __launch_bounds__(256)
void attn_topk(const unsigned short* __restrict__ ehb,   // [8192,512] bf16 (pre-scaled)
               const unsigned short* __restrict__ etb,   // [8192,512] bf16
               float* __restrict__ tw, int* __restrict__ ti) {
    const int t    = threadIdx.x;
    const int wave = t >> 5;
    const int lane = t & 31;
    const int half = lane >> 4;
    const int r16  = lane & 15;
    const int m0   = blockIdx.x * 16;

    // Hoist all 16 A-fragments (K=512) for this row tile: 128 VGPRs/lane.
    FragU a[16];
    const unsigned short* arow = ehb + (size_t)(m0 + r16) * 512;
#pragma unroll
    for (int kk = 0; kk < 16; ++kk) {
        a[kk].u[0] = *(const uint4*)&arow[kk * 32 + half * 8];
        a[kk].u[1] = *(const uint4*)&arow[kk * 32 + 16 + half * 8];
    }

    float tv[6];
    int   tix[6];
#pragma unroll
    for (int j = 0; j < 6; ++j) { tv[j] = -3.0e38f; tix[j] = 0; }

    __shared__ __align__(16) float sc[8][16][16];

    // each wave handles column tiles wave, wave+8, ... : 64 tiles each (equal trip count)
    for (int ct = wave; ct < 512; ct += 8) {
        const int n0 = ct * 16;
        v8f acc = {};
        const unsigned short* brow = etb + (size_t)(n0 + r16) * 512;
#pragma unroll
        for (int kk = 0; kk < 16; ++kk) {
            FragU b;
            b.u[0] = *(const uint4*)&brow[kk * 32 + half * 8];
            b.u[1] = *(const uint4*)&brow[kk * 32 + 16 + half * 8];
            acc = __builtin_amdgcn_wmma_f32_16x16x32_bf16(false, a[kk].v, false, b.v,
                                                          (short)0, acc, false, false);
        }
#pragma unroll
        for (int v = 0; v < 8; ++v) sc[wave][v + 8 * half][r16] = acc[v];
        __syncthreads();
        if (lane < 16) {
#pragma unroll
            for (int c = 0; c < 16; ++c) {
                float vv = sc[wave][lane][c];
                int amin = 0; float vmin = tv[0];
#pragma unroll
                for (int j = 1; j < 6; ++j) if (tv[j] < vmin) { vmin = tv[j]; amin = j; }
                if (vv > vmin) { tv[amin] = vv; tix[amin] = n0 + c; }
            }
        }
        __syncthreads();
    }

    // merge per-wave candidate lists (8 waves x 6 per row -> top-6)
    __shared__ float cv[8][16][6];
    __shared__ int   ci[8][16][6];
    if (lane < 16) {
#pragma unroll
        for (int j = 0; j < 6; ++j) { cv[wave][lane][j] = tv[j]; ci[wave][lane][j] = tix[j]; }
    }
    __syncthreads();
    if (t < 16) {
        float fv[6]; int fi[6];
#pragma unroll
        for (int j = 0; j < 6; ++j) { fv[j] = -3.0e38f; fi[j] = 0; }
        for (int w = 0; w < 8; ++w) {
#pragma unroll
            for (int j = 0; j < 6; ++j) {
                float vv = cv[w][t][j];
                int amin = 0; float vmin = fv[0];
#pragma unroll
                for (int q = 1; q < 6; ++q) if (fv[q] < vmin) { vmin = fv[q]; amin = q; }
                if (vv > vmin) { fv[amin] = vv; fi[amin] = ci[w][t][j]; }
            }
        }
#pragma unroll
        for (int j = 0; j < 6; ++j) {
            tw[(size_t)(m0 + t) * 6 + j] = fv[j];
            ti[(size_t)(m0 + t) * 6 + j] = fi[j];
        }
    }
}

// ---------------- per-node gather / gating / aggregation ----------------
__global__ __launch_bounds__(256)
void node_mix(const float* __restrict__ eh, const float* __restrict__ et,
              const float* __restrict__ tw, const int* __restrict__ ti,
              unsigned short* __restrict__ u_bf, unsigned short* __restrict__ v_bf) {
    const int n = blockIdx.x;
    const int t = threadIdx.x;

    // softmax p over the 6 logits (redundant per thread)
    float p[6]; int idx[6];
    float mx = -3.0e38f;
#pragma unroll
    for (int j = 0; j < 6; ++j) {
        p[j]   = tw[(size_t)n * 6 + j];
        idx[j] = ti[(size_t)n * 6 + j];
        mx = fmaxf(mx, p[j]);
    }
    float s = 0.f;
#pragma unroll
    for (int j = 0; j < 6; ++j) { p[j] = expf(p[j] - mx); s += p[j]; }
    float inv = 1.0f / s;
#pragma unroll
    for (int j = 0; j < 6; ++j) p[j] *= inv;

    float nb[6][2], ehd[2], ka[6] = {0, 0, 0, 0, 0, 0};
#pragma unroll
    for (int dp = 0; dp < 2; ++dp) {
        const int d = t + dp * 256;
        ehd[dp] = eh[(size_t)n * 512 + d];
#pragma unroll
        for (int j = 0; j < 6; ++j) {
            float nbv = et[(size_t)idx[j] * 512 + d];
            nb[j][dp] = nbv;
            float ehr  = p[j] * nbv + (1.0f - p[j]) * ehd[dp];
            float gate = tanhf(ehd[dp] + ehr);
            ka[j] += nbv * gate;
        }
    }

    __shared__ float red[256][6];
#pragma unroll
    for (int j = 0; j < 6; ++j) red[t][j] = ka[j];
    __syncthreads();
    for (int sr = 128; sr > 0; sr >>= 1) {
        if (t < sr) {
#pragma unroll
            for (int j = 0; j < 6; ++j) red[t][j] += red[t + sr][j];
        }
        __syncthreads();
    }

    float kp[6];
    float kmx = -3.0e38f;
#pragma unroll
    for (int j = 0; j < 6; ++j) { kp[j] = red[0][j]; kmx = fmaxf(kmx, kp[j]); }
    float ks = 0.f;
#pragma unroll
    for (int j = 0; j < 6; ++j) { kp[j] = expf(kp[j] - kmx); ks += kp[j]; }
    float kinv = 1.0f / ks;

#pragma unroll
    for (int dp = 0; dp < 2; ++dp) {
        float enh = 0.f;
#pragma unroll
        for (int j = 0; j < 6; ++j) enh += kp[j] * kinv * nb[j][dp];
        const int d = t + dp * 256;
        u_bf[(size_t)n * 512 + d] = f2bf(ehd[dp] + enh);
        v_bf[(size_t)n * 512 + d] = f2bf(ehd[dp] * enh);
    }
}

// ---------------- readout: gemv, softmax stats, pooled, finalize -----------
__global__ void att_gemv(const float* __restrict__ ath, const float* __restrict__ Wa2,
                         const float* __restrict__ ba2, float* __restrict__ att) {
    const int n = blockIdx.x * blockDim.x + threadIdx.x;   // 8192 threads
    float s = 0.f;
    const float* row = ath + (size_t)n * 256;
    for (int j = 0; j < 256; ++j) s += row[j] * Wa2[j];
    att[n] = s + ba2[0];
}

__global__ void softmax_stats(const float* __restrict__ att, float* __restrict__ stats) {
    __shared__ float rb[256];
    const int t = threadIdx.x;
    float lm = -3.0e38f;
    for (int i = t; i < 8192; i += 256) lm = fmaxf(lm, att[i]);
    rb[t] = lm; __syncthreads();
    for (int s = 128; s > 0; s >>= 1) { if (t < s) rb[t] = fmaxf(rb[t], rb[t + s]); __syncthreads(); }
    const float mx = rb[0]; __syncthreads();
    float ls = 0.f;
    for (int i = t; i < 8192; i += 256) ls += expf(att[i] - mx);
    rb[t] = ls; __syncthreads();
    for (int s = 128; s > 0; s >>= 1) { if (t < s) rb[t] += rb[t + s]; __syncthreads(); }
    if (t == 0) { stats[0] = mx; stats[1] = rb[0]; }
}

__global__ void pooled_partial(const float* __restrict__ att, const float* __restrict__ stats,
                               const float* __restrict__ emb, float* __restrict__ part) {
    __shared__ float lw[256];
    const int b = blockIdx.x, t = threadIdx.x;
    lw[t] = expf(att[b * 256 + t] - stats[0]) / stats[1];
    __syncthreads();
    float s0 = 0.f, s1 = 0.f;
    for (int r = 0; r < 256; ++r) {
        const float* row = emb + (size_t)(b * 256 + r) * 512;
        s0 += lw[r] * row[t];
        s1 += lw[r] * row[t + 256];
    }
    part[b * 512 + t]       = s0;
    part[b * 512 + t + 256] = s1;
}

__device__ __forceinline__ float block_reduce_512(float v, float* rb, int t) {
    rb[t] = v; __syncthreads();
    for (int s = 256; s > 0; s >>= 1) { if (t < s) rb[t] += rb[t + s]; __syncthreads(); }
    float r = rb[0]; __syncthreads();
    return r;
}

__global__ void finalize(const float* __restrict__ part, const float* __restrict__ gamma,
                         const float* __restrict__ beta, const float* __restrict__ Wfc,
                         const float* __restrict__ bfc, float* __restrict__ out) {
    __shared__ float rb[512];
    const int t = threadIdx.x;   // 512 threads
    float pd = 0.f;
    for (int b = 0; b < 32; ++b) pd += part[b * 512 + t];
    const float mu  = block_reduce_512(pd, rb, t) * (1.0f / 512.0f);
    const float dv  = pd - mu;
    const float var = block_reduce_512(dv * dv, rb, t) * (1.0f / 512.0f);
    const float hn  = dv * rsqrtf(var + 1e-5f) * gamma[t] + beta[t];
    const float l0  = block_reduce_512(hn * Wfc[t], rb, t);
    const float l1  = block_reduce_512(hn * Wfc[512 + t], rb, t);
    if (t == 0) { out[0] = l0 + bfc[0]; out[1] = l1 + bfc[1]; }
}

// ============================ host launcher ============================
extern "C" void kernel_launch(void* const* d_in, const int* in_sizes, int n_in,
                              void* d_out, int out_size, void* d_ws, size_t ws_size,
                              hipStream_t stream) {
    (void)in_sizes; (void)n_in; (void)out_size; (void)ws_size;
    constexpr int Nn = 8192, DIN = 1024, Dd = 512, Hh = 256;

    const float* x    = (const float*)d_in[0];
    const float* W1   = (const float*)d_in[1];
    const float* b1   = (const float*)d_in[2];
    const float* Wh   = (const float*)d_in[3];
    const float* bh   = (const float*)d_in[4];
    const float* Wt   = (const float*)d_in[5];
    const float* bt   = (const float*)d_in[6];
    const float* Wl1  = (const float*)d_in[7];
    const float* bl1  = (const float*)d_in[8];
    const float* Wl2  = (const float*)d_in[9];
    const float* bl2  = (const float*)d_in[10];
    const float* Wa1  = (const float*)d_in[11];
    const float* ba1  = (const float*)d_in[12];
    const float* Wa2  = (const float*)d_in[13];
    const float* ba2  = (const float*)d_in[14];
    const float* gamma= (const float*)d_in[15];
    const float* beta = (const float*)d_in[16];
    const float* Wfc  = (const float*)d_in[17];
    const float* bfc  = (const float*)d_in[18];

    char* w = (char*)d_ws;
    size_t off = 0;
    auto alloc = [&](size_t bytes) { size_t o = off; off = (off + bytes + 255) & ~(size_t)255; return o; };
    const size_t oXBF  = alloc((size_t)Nn * DIN * 2);   // x bf16   | later: e_t fp32 (16MB)
    const size_t oW1B  = alloc((size_t)Dd * DIN * 2);
    const size_t oWHB  = alloc((size_t)Dd * Dd * 2);
    const size_t oWTB  = alloc((size_t)Dd * Dd * 2);
    const size_t oWL1B = alloc((size_t)Dd * Dd * 2);
    const size_t oWL2B = alloc((size_t)Dd * Dd * 2);
    const size_t oWA1B = alloc((size_t)Hh * Dd * 2);
    const size_t oHPRE = alloc((size_t)Nn * Dd * 4);    // h_pre    | later: emb fp32
    const size_t oCP   = alloc((size_t)32 * Dd * 4);
    const size_t oCM   = alloc((size_t)Dd * 4);
    const size_t oHBF  = alloc((size_t)Nn * Dd * 2);    // h bf16   | later: u_bf
    const size_t oEH   = alloc((size_t)Nn * Dd * 4);    // e_h fp32 | later: att_h
    const size_t oEHB  = alloc((size_t)Nn * Dd * 2);    // scaled e_h bf16 | later: v_bf
    const size_t oETB  = alloc((size_t)Nn * Dd * 2);    // e_t bf16 | later: emb bf16
    const size_t oTW   = alloc((size_t)Nn * 6 * 4);
    const size_t oTI   = alloc((size_t)Nn * 6 * 4);
    const size_t oATT  = alloc((size_t)Nn * 4);
    const size_t oST   = alloc(2 * 4);
    const size_t oPP   = alloc((size_t)32 * Dd * 4);

    unsigned short* XBF  = (unsigned short*)(w + oXBF);
    unsigned short* W1B  = (unsigned short*)(w + oW1B);
    unsigned short* WHB  = (unsigned short*)(w + oWHB);
    unsigned short* WTB  = (unsigned short*)(w + oWTB);
    unsigned short* WL1B = (unsigned short*)(w + oWL1B);
    unsigned short* WL2B = (unsigned short*)(w + oWL2B);
    unsigned short* WA1B = (unsigned short*)(w + oWA1B);
    float*          HPRE = (float*)(w + oHPRE);
    float*          EMB  = HPRE;                         // alias
    float*          CP   = (float*)(w + oCP);
    float*          CM   = (float*)(w + oCM);
    unsigned short* HBF  = (unsigned short*)(w + oHBF);
    unsigned short* UBF  = HBF;                          // alias
    float*          EH   = (float*)(w + oEH);
    float*          ATTH = EH;                           // alias (8MB fits in 16MB)
    unsigned short* EHB  = (unsigned short*)(w + oEHB);
    unsigned short* VBF  = EHB;                          // alias
    unsigned short* ETB  = (unsigned short*)(w + oETB);
    unsigned short* EMBB = ETB;                          // alias
    float*          ETF  = (float*)(w + oXBF);           // alias (16MB == 16MB)
    float*          TW   = (float*)(w + oTW);
    int*            TI   = (int*)(w + oTI);
    float*          ATT  = (float*)(w + oATT);
    float*          ST   = (float*)(w + oST);
    float*          PP   = (float*)(w + oPP);

    const float scale = 0.044194173824159216f;  // 512^-0.5

    // 1) bf16 conversions of x + all weight matrices
    cvt_f32_bf16<<<(Nn * DIN) / 256, 256, 0, stream>>>(x,   XBF,  Nn * DIN);
    cvt_f32_bf16<<<(Dd * DIN) / 256, 256, 0, stream>>>(W1,  W1B,  Dd * DIN);
    cvt_f32_bf16<<<(Dd * Dd)  / 256, 256, 0, stream>>>(Wh,  WHB,  Dd * Dd);
    cvt_f32_bf16<<<(Dd * Dd)  / 256, 256, 0, stream>>>(Wt,  WTB,  Dd * Dd);
    cvt_f32_bf16<<<(Dd * Dd)  / 256, 256, 0, stream>>>(Wl1, WL1B, Dd * Dd);
    cvt_f32_bf16<<<(Dd * Dd)  / 256, 256, 0, stream>>>(Wl2, WL2B, Dd * Dd);
    cvt_f32_bf16<<<(Hh * Dd)  / 256, 256, 0, stream>>>(Wa1, WA1B, Hh * Dd);

    // 2) h_pre = lrelu(x @ W1^T + b1)
    gemm_bf16_wmma<<<dim3(Dd / 64, Nn / 64), 256, 0, stream>>>(
        XBF, W1B, b1, HPRE, nullptr, 1.f, Nn, Dd, DIN, GF_LRELU);

    // 3) h = (h_pre + mean_N(h_pre)) * 0.5  -> bf16
    colsum_partial<<<32, 256, 0, stream>>>(HPRE, CP);
    colmean_final<<<1, 512, 0, stream>>>(CP, CM);
    mix_mean_bf16<<<(Nn * Dd) / 256, 256, 0, stream>>>(HPRE, CM, HBF, Nn * Dd);

    // 4) e_h (fp32 + pre-scaled bf16), e_t (fp32 + bf16)
    gemm_bf16_wmma<<<dim3(Dd / 64, Nn / 64), 256, 0, stream>>>(
        HBF, WHB, bh, EH,  EHB, scale, Nn, Dd, Dd, 0);
    gemm_bf16_wmma<<<dim3(Dd / 64, Nn / 64), 256, 0, stream>>>(
        HBF, WTB, bt, ETF, ETB, 1.f,   Nn, Dd, Dd, 0);

    // 5) streaming attention logits + per-row top-6 (never materializes 8192x8192)
    attn_topk<<<Nn / 16, 256, 0, stream>>>(EHB, ETB, TW, TI);

    // 6) gather + gating + KNN aggregation -> u = e_h+e_Nh, v = e_h*e_Nh (bf16)
    node_mix<<<Nn, 256, 0, stream>>>(EH, ETF, TW, TI, UBF, VBF);

    // 7) emb = lrelu(u@Wl1^T+bl1) + lrelu(v@Wl2^T+bl2)   (fp32 + bf16 copy)
    gemm_bf16_wmma<<<dim3(Dd / 64, Nn / 64), 256, 0, stream>>>(
        UBF, WL1B, bl1, EMB, nullptr, 1.f, Nn, Dd, Dd, GF_LRELU);
    gemm_bf16_wmma<<<dim3(Dd / 64, Nn / 64), 256, 0, stream>>>(
        VBF, WL2B, bl2, EMB, EMBB,    1.f, Nn, Dd, Dd, GF_LRELU | GF_ACC);

    // 8) att_h = lrelu(emb @ Wa1^T + ba1)   [8192,256]
    gemm_bf16_wmma<<<dim3(Hh / 64, Nn / 64), 256, 0, stream>>>(
        EMBB, WA1B, ba1, ATTH, nullptr, 1.f, Nn, Hh, Dd, GF_LRELU);

    // 9) att[n] = att_h[n,:] . Wa2 + ba2 ; softmax over nodes ; pooled ; layernorm+fc
    att_gemv<<<Nn / 256, 256, 0, stream>>>(ATTH, Wa2, ba2, ATT);
    softmax_stats<<<1, 256, 0, stream>>>(ATT, ST);
    pooled_partial<<<32, 256, 0, stream>>>(ATT, ST, EMB, PP);
    finalize<<<1, 512, 0, stream>>>(PP, gamma, beta, Wfc, bfc, (float*)d_out);
}
